// HashedImageField_9285719294006
// MI455X (gfx1250) — matrix operand order repaired and checked by reading, not compiled, not run on hardware
//
#include <hip/hip_runtime.h>

typedef __attribute__((ext_vector_type(16))) _Float16 v16h;
typedef __attribute__((ext_vector_type(8)))  float    v8f;

#define NPB     128             // points per block = 4 wave32s; each wave owns 2 M-tiles
#define HASHSZ  524288u         // 2^19 entries per level
#define LDF     40              // feature row stride (f16): 80B rows (aliases 20-f32 partial rows)
#define LDH     72              // hidden  row stride (f16): 144B rows, 16B-aligned, 36 dw (gcd 4)
#define LRED    20              // f32 partial row stride (byte-identical to LDF rows)
#define NFRAG   12              // 4 (W0) + 4 (W1, phys K 0..31) + 4 (W1, phys K 32..63)
#define FRAG_ELEMS (32 * 16)    // f16 per fragment (32 lanes x 16 values)

union FragU { uint4 q[2]; v16h v; };

__device__ __forceinline__ v16h ldq2(const _Float16* p) {   // 16 contiguous f16 -> fragment
  FragU u;
  u.q[0] = *(const uint4*)p;
  u.q[1] = *(const uint4*)(p + 8);
  return u.v;
}

// Single-instruction ReLU: signed-int view of IEEE f32 is monotonic; negatives
// (incl. -0) have negative int repr => v_max_i32 with 0 == relu. No canonicalize.
__device__ __forceinline__ float relu1(float v) {
  return __int_as_float(max(__float_as_int(v), 0));
}

// K-permutation baked into h1 storage and W1 fragments: physical position pos holds
// logical K  sigma(pos) = (pos&3)*16 + (pos>>2).  (Bijection on 0..63; makes each
// lane's 4 layer-1 outputs contiguous while keeping A-fragment runs contiguous.)
__global__ void prep_weight_frags(const float* __restrict__ W0,
                                  const float* __restrict__ W1,
                                  _Float16* __restrict__ wsF) {
  const int i = blockIdx.x * blockDim.x + threadIdx.x;
  if (i >= NFRAG * FRAG_ELEMS) return;
  const int f = i >> 9;              // fragment id 0..11
  const int l = (i >> 4) & 31;       // lane
  const int j = i & 15;              // element within lane
  const int col = (f & 3) * 16 + (l & 15);
  const int hl  = (l >> 4) & 1;
  float v;
  if (f < 4) {                       // layer-1 weights: unpermuted, K>=16 is zero pad
    const int k = (j < 8) ? hl * 8 + j : 16 + hl * 8 + (j - 8);
    v = (k < 16) ? W0[k * 64 + col] : 0.f;
  } else {                           // layer-2 weights: hardware-K pos -> logical sigma(pos)
    const int pos  = ((f >= 8) ? 32 : 0) + ((j < 8) ? hl * 8 + j : 16 + hl * 8 + (j - 8));
    const int klog = (pos & 3) * 16 + (pos >> 2);
    v = W1[klog * 64 + col];
  }
  wsF[i] = (_Float16)v;
}

__global__ __launch_bounds__(NPB) void ngp_hashfield_fused(
    const float* __restrict__ x,       // (N,3)
    const float* __restrict__ extent,  // (1,)
    const float* __restrict__ tables,  // (8, 2^19, 2) f32
    const float* __restrict__ W2,      // (64,1)
    const _Float16* __restrict__ wsF,  // precomputed B fragments
    float* __restrict__ out)           // (N,)
{
  // level constants: scale = 8*b^l - 1, b = 20^(1/7); res = ceil(scale)+1
  constexpr float kScale[8] = {7.0f, 11.272984f, 17.828268f, 27.884864f,
                               43.313008f, 66.981640f, 103.292240f, 159.0f};
  constexpr int   kRes[8]   = {8, 13, 19, 29, 45, 68, 105, 160};

  __shared__ __align__(16) _Float16 sF[NPB * LDF];   // features; aliased as f32 partials
  __shared__ __align__(16) _Float16 sH[NPB * LDH];   // h1, K-permuted physical layout
  float* const sRed = (float*)sF;

  const int tid  = threadIdx.x;
  const int lane = tid & 31;
  const int wave = tid >> 5;
  const int c16  = lane & 15;
  const int hl   = (lane >> 4) & 1;
  const int p    = blockIdx.x * NPB + tid;

  // ---- loop-invariant weight fragments (L2-hot, 24x global b128) ----
  const _Float16* wf = wsF + lane * 16;
  v16h b0[4], b1l[4], b1h[4];
#pragma unroll
  for (int t = 0; t < 4; ++t) {
    b0[t]  = ldq2(wf + (t    ) * FRAG_ELEMS);
    b1l[t] = ldq2(wf + (4 + t) * FRAG_ELEMS);
    b1h[t] = ldq2(wf + (8 + t) * FRAG_ELEMS);
  }
  float w2c[4];
#pragma unroll
  for (int t = 0; t < 4; ++t) w2c[t] = W2[t * 16 + c16];

  // ---------------- hash-grid encode: one point per lane ----------------
  const float inv_ext = 1.0f / extent[0];
  const float xn0 = x[p * 3 + 0] * inv_ext;
  const float xn1 = x[p * 3 + 1] * inv_ext;
  const float xn2 = x[p * 3 + 2] * inv_ext;

  unsigned* const frow = (unsigned*)(sF + tid * LDF);   // 8x b32 -> compiler merges to b128
#pragma unroll
  for (int l = 0; l < 8; ++l) {
    const float s  = kScale[l];
    const int   rs = kRes[l];
    const float px = xn0 * s + 0.5f, py = xn1 * s + 0.5f, pz = xn2 * s + 0.5f;
    const float gx = floorf(px), gy = floorf(py), gz = floorf(pz);
    const float tx = px - gx, ty = py - gy, tz = pz - gz;
    const int ix = (int)gx, iy = (int)gy, iz = (int)gz;
    const float wx0 = 1.f - tx, wy0 = 1.f - ty, wz0 = 1.f - tz;
    const float w00 = wx0 * wy0, w01 = wx0 * ty, w10 = tx * wy0, w11 = tx * ty;
    const float2* t2 = (const float2*)(tables + (size_t)l * (HASHSZ * 2u));
    float f0 = 0.f, f1 = 0.f;

    if (l < 6) {
      // dense level: pos >= 0.5 -> base >= 0; upper clamp folded into the +1 terms
      const int X0 = ix,            X1 = min(ix + 1, rs - 1);
      const int Y0 = iy * rs,       Y1 = min(iy + 1, rs - 1) * rs;
      const int Z0 = iz * rs * rs,  Z1 = min(iz + 1, rs - 1) * rs * rs;
#pragma unroll
      for (int c = 0; c < 8; ++c) {
        const int cx = (c >> 2) & 1, cy = (c >> 1) & 1, cz = c & 1;
        const float w = (cx ? (cy ? w11 : w10) : (cy ? w01 : w00)) * (cz ? tz : wz0);
        const unsigned idx = (unsigned)((cx ? X1 : X0) + (cy ? Y1 : Y0) + (cz ? Z1 : Z0));
        const float2 v = t2[idx];
        f0 += w * v.x;
        f1 += w * v.y;
      }
    } else {
      const unsigned hx0 = (unsigned)ix,               hx1 = hx0 + 1u;
      const unsigned hy0 = (unsigned)iy * 2654435761u, hy1 = hy0 + 2654435761u;
      const unsigned hz0 = (unsigned)iz * 805459861u,  hz1 = hz0 + 805459861u;
#pragma unroll
      for (int c = 0; c < 8; ++c) {
        const int cx = (c >> 2) & 1, cy = (c >> 1) & 1, cz = c & 1;
        const float w = (cx ? (cy ? w11 : w10) : (cy ? w01 : w00)) * (cz ? tz : wz0);
        const unsigned idx = ((cx ? hx1 : hx0) ^ (cy ? hy1 : hy0) ^ (cz ? hz1 : hz0))
                             & (HASHSZ - 1u);
        const float2 v = t2[idx];
        f0 += w * v.x;
        f1 += w * v.y;
      }
    }
    union { _Float16 h[2]; unsigned u; } pk;
    pk.h[0] = (_Float16)f0;
    pk.h[1] = (_Float16)f1;
    frow[l] = pk.u;
  }
  // No zero-pad stores: layer-1 A top half is supplied as register zeros.
  // No __syncthreads anywhere: all LDS rows are wave-local, wave LDS ops are in-order.

  // ---------------- MLP via WMMA: each wave owns its own 2 x 16-row tiles ----------------
#pragma unroll
  for (int mt = 0; mt < 2; ++mt) {
    const int row0 = wave * 32 + mt * 16;
    v8f acc[4];

    // layer 1: feats(16x16) @ W0 (K padded to 32; pad half = register zeros x zero B rows)
    {
      const _Float16* af = sF + (row0 + c16) * LDF + hl * 8;
      FragU u;
      u.q[0] = *(const uint4*)af;                 // K = 8*hl .. +7 (real features)
      const uint4 z = {0u, 0u, 0u, 0u};
      u.q[1] = z;                                 // K = 16.. : zeros (B rows are zero too)
#pragma unroll
      for (int t = 0; t < 4; ++t) {
        v8f cz = {};
        acc[t] = __builtin_amdgcn_wmma_f32_16x16x32_f16(false, u.v, false, b0[t],
                                                        (short)0, cz, false, false);
      }
    }

    // relu -> f16 -> sH in K-permuted layout: lane's 4 outputs per row are contiguous
    {
      _Float16* hp = sH + (row0 + hl * 8) * LDH + c16 * 4;
#pragma unroll
      for (int i = 0; i < 8; ++i) {
        union { _Float16 h[4]; uint2 u; } pk4;
        pk4.h[0] = (_Float16)relu1(acc[0][i]);    // logical col c16      -> phys 4*c16+0
        pk4.h[1] = (_Float16)relu1(acc[1][i]);    // logical col c16+16   -> phys 4*c16+1
        pk4.h[2] = (_Float16)relu1(acc[2][i]);    // logical col c16+32   -> phys 4*c16+2
        pk4.h[3] = (_Float16)relu1(acc[3][i]);    // logical col c16+48   -> phys 4*c16+3
        *(uint2*)(hp + i * LDH) = pk4.u;          // one b64 store per row
      }
    }

    // layer 2: h1(16x64) @ W1 (K-permuted consistently in A and B), two K=32 steps
    {
      const _Float16* ah = sH + (row0 + c16) * LDH + hl * 8;
      FragU a0, a1;
      a0.q[0] = *(const uint4*)(ah);
      a0.q[1] = *(const uint4*)(ah + 16);
      a1.q[0] = *(const uint4*)(ah + 32);
      a1.q[1] = *(const uint4*)(ah + 48);
#pragma unroll
      for (int t = 0; t < 4; ++t) {
        v8f cz = {};
        v8f c = __builtin_amdgcn_wmma_f32_16x16x32_f16(false, a0.v, false, b1l[t],
                                                       (short)0, cz, false, false);
        acc[t] = __builtin_amdgcn_wmma_f32_16x16x32_f16(false, a1.v, false, b1h[t],
                                                        (short)0, c, false, false);
      }
    }

    // fused layer 3 partial: relu(h2) . W2 straight from C-layout registers
    {
      float* rp = sRed + (row0 + hl * 8) * LRED + c16;
#pragma unroll
      for (int i = 0; i < 8; ++i) {
        float sum = 0.f;
#pragma unroll
        for (int t = 0; t < 4; ++t) sum += relu1(acc[t][i]) * w2c[t];
        rp[i * LRED] = sum;
      }
    }
  }

  // ---------------- reduce 16 partials per point (wave-local rows, LDS in-order) --------
  const float4* rr = (const float4*)(sRed + tid * LRED);
  const float4 r0 = rr[0], r1 = rr[1], r2 = rr[2], r3 = rr[3];
  out[p] = ((r0.x + r0.y) + (r0.z + r0.w)) + ((r1.x + r1.y) + (r1.z + r1.w)) +
           ((r2.x + r2.y) + (r2.z + r2.w)) + ((r3.x + r3.y) + (r3.z + r3.w));
}

extern "C" void kernel_launch(void* const* d_in, const int* in_sizes, int n_in,
                              void* d_out, int out_size, void* d_ws, size_t ws_size,
                              hipStream_t stream) {
  const float* x   = (const float*)d_in[0];
  const float* ext = (const float*)d_in[1];
  const float* tbl = (const float*)d_in[2];
  const float* W0  = (const float*)d_in[3];
  const float* W1  = (const float*)d_in[4];
  const float* W2  = (const float*)d_in[5];
  float* out = (float*)d_out;
  _Float16* wsF = (_Float16*)d_ws;        // 12 KB of workspace for weight fragments

  prep_weight_frags<<<(NFRAG * FRAG_ELEMS + 127) / 128, 128, 0, stream>>>(W0, W1, wsF);

  const int n = in_sizes[0] / 3;          // 2,097,152 points (divisible by NPB)
  ngp_hashfield_fused<<<n / NPB, NPB, 0, stream>>>(x, ext, tbl, W2, wsF, out);
}
